// ExplainableHeteroClassifier_23742579212721
// MI455X (gfx1250) — matrix operand order repaired and checked by reading, not compiled, not run on hardware
//
#include <hip/hip_runtime.h>
#include <hip/hip_bf16.h>
#include <math.h>

// ---------------------------------------------------------------------------
// ExplainableHeteroClassifier for MI455X (gfx1250, wave32, WMMA).
// GEMMs: 128x128x32-tiled bf16 WMMA (v_wmma_f32_16x16x32_bf16), f32 accum,
// double-buffered LDS ping-pong + register prefetch (1 barrier / K-step).
// fp32->bf16 conversion = add-half + v_perm_b32 pack (3 VALU per 2 elems).
// Attention softmax/scatter use global atomics (negligible work fraction).
// ---------------------------------------------------------------------------

typedef __attribute__((ext_vector_type(16))) __bf16          v16bf;
typedef __attribute__((ext_vector_type(8)))  float           v8f;
typedef __attribute__((ext_vector_type(8)))  unsigned short  ushort8;

union FragU { ushort8 h[2]; v16bf v; };

// pack 2 floats -> 2 bf16 (round half-up) in one dword: one v_perm_b32 each
__device__ __forceinline__ unsigned int pack2bf(float x, float y) {
  unsigned int ux = __float_as_uint(x) + 0x8000u;
  unsigned int uy = __float_as_uint(y) + 0x8000u;
  // {src0=uy (bytes 4-7), src1=ux (bytes 0-3)}: take x[31:16] -> lo, y[31:16] -> hi
  return __builtin_amdgcn_perm(uy, ux, 0x07060302u);
}
__device__ __forceinline__ uint2 pack4bf(float x, float y, float z, float w) {
  uint2 r;
  r.x = pack2bf(x, y);
  r.y = pack2bf(z, w);
  return r;
}

#define BM 128
#define BN 128
#define BK 32
#define LDP 40   // padded LDS row stride (half-words): 80B rows -> 8/16B-aligned ops

// C[M,N] = A[M,K] @ B[K,N] (+ bias row), bf16 WMMA, f32 accum, ping-pong LDS.
template <int HASB>
__global__ __launch_bounds__(256) void gemm_bf16_wmma(
    const float* __restrict__ A, const float* __restrict__ B,
    float* __restrict__ C, const float* __restrict__ bias,
    int M, int N, int K)
{
  __shared__ unsigned short sA[2][BM * LDP];  // [m][k]
  __shared__ unsigned short sB[2][BN * LDP];  // [n][k]  (B transposed)

  const int tid  = threadIdx.x;
  const int wave = tid >> 5, lane = tid & 31;
  const int wm = wave & 3, wn = wave >> 2;     // 4x2 wave grid -> 32x64 per wave
  const int l15 = lane & 15, half = lane >> 4;
  const int bm = blockIdx.y * BM, bn = blockIdx.x * BN;

  // staging coordinates (fixed per thread)
  const int ar_row = tid >> 3;                 // A: rows advance by 32 per chunk
  const int ar_col = (tid & 7) << 2;
  const int bn_col = tid & 127;                // B: column within tile
  const int bk_q   = (tid >> 7) << 2;          //    k-quad base (0 or 4), +8 per chunk

  v8f acc[2][4];
#pragma unroll
  for (int i = 0; i < 2; ++i)
#pragma unroll
    for (int j = 0; j < 4; ++j)
#pragma unroll
      for (int r = 0; r < 8; ++r) acc[i][j][r] = 0.0f;

  float4 ra[4];          // A prefetch: 4 float4 row-chunks
  float  rb[4][4];       // B prefetch: 4 k-quads of one column each

  // ---- prologue: prefetch tile k0 = 0 ----
#pragma unroll
  for (int i = 0; i < 4; ++i)
    ra[i] = *(const float4*)(A + (size_t)(bm + ar_row + (i << 5)) * K + ar_col);
#pragma unroll
  for (int i = 0; i < 4; ++i) {
    int kq = bk_q + (i << 3);
#pragma unroll
    for (int j = 0; j < 4; ++j)
      rb[i][j] = B[(size_t)(kq + j) * N + (bn + bn_col)];
  }

  int buf = 0;
  for (int k0 = 0; k0 < K; k0 += BK, buf ^= 1) {
    // ---- commit prefetched tile to LDS[buf] (bf16, packed b64 stores) ----
#pragma unroll
    for (int i = 0; i < 4; ++i) {
      uint2 p = pack4bf(ra[i].x, ra[i].y, ra[i].z, ra[i].w);
      *(uint2*)&sA[buf][(ar_row + (i << 5)) * LDP + ar_col] = p;
    }
#pragma unroll
    for (int i = 0; i < 4; ++i) {
      uint2 p = pack4bf(rb[i][0], rb[i][1], rb[i][2], rb[i][3]);
      *(uint2*)&sB[buf][bn_col * LDP + bk_q + (i << 3)] = p;
    }
    __syncthreads();

    // ---- issue next tile's global loads early (latency hides under WMMA) ----
    if (k0 + BK < K) {
      const int kn = k0 + BK;
#pragma unroll
      for (int i = 0; i < 4; ++i)
        ra[i] = *(const float4*)(A + (size_t)(bm + ar_row + (i << 5)) * K + kn + ar_col);
#pragma unroll
      for (int i = 0; i < 4; ++i) {
        int kq = kn + bk_q + (i << 3);
#pragma unroll
        for (int j = 0; j < 4; ++j)
          rb[i][j] = B[(size_t)(kq + j) * N + (bn + bn_col)];
      }
    }

    // ---- fragments + WMMA ----
    FragU a[2], b[4];
#pragma unroll
    for (int i = 0; i < 2; ++i) {        // A frag: lane m=l15; K = 8h+0..7, 16+8h+0..7
      int row = (wm << 5) + (i << 4) + l15;
      a[i].h[0] = *(const ushort8*)&sA[buf][row * LDP + (half << 3)];
      a[i].h[1] = *(const ushort8*)&sA[buf][row * LDP + 16 + (half << 3)];
    }
#pragma unroll
    for (int j = 0; j < 4; ++j) {        // B frag: lane n=l15; K = 8h+0..7, 16+8h+0..7
      int cn = (wn << 6) + (j << 4) + l15;
      b[j].h[0] = *(const ushort8*)&sB[buf][cn * LDP + (half << 3)];
      b[j].h[1] = *(const ushort8*)&sB[buf][cn * LDP + 16 + (half << 3)];
    }
#pragma unroll
    for (int i = 0; i < 2; ++i)
#pragma unroll
      for (int j = 0; j < 4; ++j)
        acc[i][j] = __builtin_amdgcn_wmma_f32_16x16x32_bf16(
            false, a[i].v, false, b[j].v, (short)0, acc[i][j], false, false);
  }

#pragma unroll
  for (int i = 0; i < 2; ++i)
#pragma unroll
    for (int j = 0; j < 4; ++j) {
      int n = bn + (wn << 6) + (j << 4) + l15;
      float bv = HASB ? bias[n] : 0.0f;
#pragma unroll
      for (int r = 0; r < 8; ++r) {      // C/D: lane n=l15, m = r + 8*half
        int m = bm + (wm << 5) + (i << 4) + (half << 3) + r;
        C[(size_t)m * N + n] = acc[i][j][r] + bv;
      }
    }
}

// ---------------------------------------------------------------------------
// out[i] = a[i] + a[i+n] + a[i+2n] + a[i+3n]   (sum over R=4 relations)
__global__ void sum4_kernel(const float* __restrict__ a, float* __restrict__ out, size_t n) {
  size_t i = (size_t)blockIdx.x * blockDim.x + threadIdx.x;
  if (i < n) out[i] = a[i] + a[i + n] + a[i + 2 * n] + a[i + 3 * n];
}

// el/er: per node n, per head h: sum_o feat[n,h,o]*a{l,r}[h,o]
__global__ __launch_bounds__(256) void el_er_kernel(
    const float* __restrict__ feat, const float* __restrict__ al,
    const float* __restrict__ ar, float* __restrict__ el, float* __restrict__ er)
{
  int n = blockIdx.x, tid = threadIdx.x;
  int h = tid >> 6, j = tid & 63;
  const float* f = feat + (size_t)n * 4096 + h * 1024;
  const float* pl = al + h * 1024;
  const float* pr = ar + h * 1024;
  float sl = 0.f, sr = 0.f;
  for (int o = j; o < 1024; o += 64) { float fv = f[o]; sl += fv * pl[o]; sr += fv * pr[o]; }
  __shared__ float bl[256], br[256];
  bl[tid] = sl; br[tid] = sr;
  __syncthreads();
  for (int s = 32; s > 0; s >>= 1) {
    if (j < s) { bl[tid] += bl[tid + s]; br[tid] += br[tid + s]; }
    __syncthreads();
  }
  if (j == 0) { el[n * 4 + h] = bl[tid]; er[n * 4 + h] = br[tid]; }
}

__device__ __forceinline__ void atomicMaxF(float* addr, float val) {
  unsigned int* ua = (unsigned int*)addr;
  unsigned int old = *ua;
  while (__uint_as_float(old) < val) {
    unsigned int assumed = old;
    old = atomicCAS(ua, assumed, __float_as_uint(val));
    if (old == assumed) break;
  }
}

__global__ void init_ms_kernel(float* m, float* s, int n) {
  int i = blockIdx.x * blockDim.x + threadIdx.x;
  if (i < n) { m[i] = -INFINITY; s[i] = 0.f; }
}

// e = leaky_relu(el[src]+er[dst], 0.2); segment-max into m[dst]
__global__ void edge_logits_kernel(const int* __restrict__ src, const int* __restrict__ dst,
                                   const float* __restrict__ el, const float* __restrict__ er,
                                   float* __restrict__ ee, float* __restrict__ m, int E) {
  int i = blockIdx.x * blockDim.x + threadIdx.x;
  if (i >= E * 4) return;
  int e = i >> 2, h = i & 3;
  int s = src[e], d = dst[e];
  float v = el[s * 4 + h] + er[d * 4 + h];
  v = v >= 0.f ? v : 0.2f * v;
  ee[i] = v;
  atomicMaxF(&m[d * 4 + h], v);
}

__global__ void fix_m_kernel(float* m, int n) {
  int i = blockIdx.x * blockDim.x + threadIdx.x;
  if (i < n) { float v = m[i]; if (!isfinite(v)) m[i] = 0.f; }
}

// p = exp(e - m[dst]); segment-sum into s[dst]
__global__ void edge_exp_kernel(const int* __restrict__ dst, const float* __restrict__ ee,
                                const float* __restrict__ m, float* __restrict__ s, int E) {
  int i = blockIdx.x * blockDim.x + threadIdx.x;
  if (i >= E * 4) return;
  int e = i >> 2, h = i & 3;
  int d = dst[e];
  atomicAdd(&s[d * 4 + h], expf(ee[i] - m[d * 4 + h]));
}

// acc[dst, h, :] += alpha * feat[src, h, :]   (one block per edge*head)
__global__ __launch_bounds__(256) void edge_scatter_kernel(
    const int* __restrict__ src, const int* __restrict__ dst,
    const float* __restrict__ ee, const float* __restrict__ m,
    const float* __restrict__ s, const float* __restrict__ feat,
    float* __restrict__ acc, int E)
{
  int eh = blockIdx.x;
  int e = eh >> 2, h = eh & 3;
  int sv = src[e], d = dst[e];
  float den = s[d * 4 + h];
  float alpha = expf(ee[eh] - m[d * 4 + h]) / (den > 0.f ? den : 1.f);
  const float* f = feat + (size_t)sv * 4096 + h * 1024;
  float* a = acc + (size_t)d * 4096 + h * 1024;
  for (int o = threadIdx.x; o < 1024; o += 256)
    atomicAdd(&a[o], alpha * f[o]);
}

__global__ void elu_div4_kernel(const float* __restrict__ acc, float* __restrict__ h, size_t n) {
  size_t i = (size_t)blockIdx.x * blockDim.x + threadIdx.x;
  if (i < n) { float x = acc[i] * 0.25f; h[i] = x > 0.f ? x : expm1f(x); }
}

// layer-2 accumulator init: 4*h (identity residual over 4 relations) + sum_r b2
__global__ void init_acc2_kernel(const float* __restrict__ h, const float* __restrict__ bsum,
                                 float* __restrict__ acc, size_t n) {
  size_t i = (size_t)blockIdx.x * blockDim.x + threadIdx.x;
  if (i < n) acc[i] = 4.f * h[i] + bsum[i & 4095];
}

// repr[c] = mean_n acc[n,c] / 4   (coalesced: consecutive threads = consecutive cols)
__global__ void col_mean_kernel(const float* __restrict__ acc, float* __restrict__ repr, int Nv) {
  int c = blockIdx.x * blockDim.x + threadIdx.x;
  if (c < 4096) {
    float s = 0.f;
    for (int n = 0; n < Nv; ++n) s += acc[(size_t)n * 4096 + c];
    repr[c] = s * (0.25f / (float)Nv);
  }
}

// y[j] = act(dot(x, W[:,j]) + b[j]);  thread-per-output, coalesced W reads
__global__ __launch_bounds__(256) void gemv_kernel(
    const float* __restrict__ x, const float* __restrict__ W,
    const float* __restrict__ b, float* __restrict__ y, int K, int Nout, int relu) {
  int j = blockIdx.x * 256 + threadIdx.x;
  if (j >= Nout) return;
  float s = b[j];
  for (int k = 0; k < K; ++k) s = fmaf(x[k], W[(size_t)k * Nout + j], s);
  y[j] = relu ? fmaxf(s, 0.f) : s;
}

// ---------------------------------------------------------------------------
extern "C" void kernel_launch(void* const* d_in, const int* in_sizes, int n_in,
                              void* d_out, int out_size, void* d_ws, size_t ws_size,
                              hipStream_t stream) {
  (void)in_sizes; (void)n_in; (void)out_size; (void)ws_size;

  const float* xg[3]   = {(const float*)d_in[0], (const float*)d_in[1], (const float*)d_in[2]};
  const int*   srcg[3] = {(const int*)d_in[3], (const int*)d_in[5], (const int*)d_in[7]};
  const int*   dstg[3] = {(const int*)d_in[4], (const int*)d_in[6], (const int*)d_in[8]};
  const float* W1  = (const float*)d_in[9];
  const float* al1 = (const float*)d_in[10];
  const float* ar1 = (const float*)d_in[11];
  const float* b1  = (const float*)d_in[12];
  const float* res1= (const float*)d_in[13];
  const float* W2  = (const float*)d_in[14];
  const float* al2 = (const float*)d_in[15];
  const float* ar2 = (const float*)d_in[16];
  const float* b2  = (const float*)d_in[17];
  const float* Wp  = (const float*)d_in[18];
  const float* bp  = (const float*)d_in[19];
  const float* Wc1 = (const float*)d_in[20];
  const float* bc1 = (const float*)d_in[21];
  const float* Wc2 = (const float*)d_in[22];
  const float* bc2 = (const float*)d_in[23];

  const int Nv_[3] = {1024, 1024, 2048};
  const int Ev_[3] = {8192, 8192, 16384};

  float* ws = (float*)d_ws;
  size_t off = 0;
  auto alloc = [&](size_t nelem) { float* p = ws + off; off += nelem; return p; };
  float* feat   = alloc((size_t)2048 * 4096);
  float* acc    = alloc((size_t)2048 * 4096);
  float* hbuf   = alloc((size_t)2048 * 4096);
  float* ressum = alloc((size_t)1024 * 4096);
  float* elb    = alloc(2048 * 4);
  float* erb    = alloc(2048 * 4);
  float* mb     = alloc(2048 * 4);
  float* sbuf   = alloc(2048 * 4);
  float* eedge  = alloc((size_t)16384 * 4);
  float* bsum1  = alloc(4096);
  float* bsum2  = alloc(4096);
  float* repr   = alloc(3 * 4096);
  float* zb     = alloc(3072);
  float* hcb    = alloc(1024);

  // pre-sum relation-shared terms: sum_r res1[r], sum_r b1[r], sum_r b2[r]
  {
    size_t n = (size_t)1024 * 4096;
    sum4_kernel<<<(unsigned)((n + 255) / 256), 256, 0, stream>>>(res1, ressum, n);
    sum4_kernel<<<16, 256, 0, stream>>>(b1, bsum1, 4096);
    sum4_kernel<<<16, 256, 0, stream>>>(b2, bsum2, 4096);
  }

  for (int g = 0; g < 3; ++g) {
    const int Nv = Nv_[g], E = Ev_[g];
    const dim3 ggrid(4096 / BN, Nv / BM);

    // ----- layer 1: acc = x @ (sum_r res1) + sum_r b1, then attention scatter -----
    gemm_bf16_wmma<1><<<ggrid, 256, 0, stream>>>(xg[g], ressum, acc, bsum1, Nv, 4096, 1024);
    for (int r = 0; r < 4; ++r) {
      const int* sp = srcg[g] + (size_t)r * E;
      const int* dp = dstg[g] + (size_t)r * E;
      gemm_bf16_wmma<0><<<ggrid, 256, 0, stream>>>(xg[g], W1 + (size_t)r * 1024 * 4096,
                                                   feat, nullptr, Nv, 4096, 1024);
      el_er_kernel<<<Nv, 256, 0, stream>>>(feat, al1 + (size_t)r * 4096, ar1 + (size_t)r * 4096, elb, erb);
      init_ms_kernel<<<(Nv * 4 + 255) / 256, 256, 0, stream>>>(mb, sbuf, Nv * 4);
      edge_logits_kernel<<<(E * 4 + 255) / 256, 256, 0, stream>>>(sp, dp, elb, erb, eedge, mb, E);
      fix_m_kernel<<<(Nv * 4 + 255) / 256, 256, 0, stream>>>(mb, Nv * 4);
      edge_exp_kernel<<<(E * 4 + 255) / 256, 256, 0, stream>>>(dp, eedge, mb, sbuf, E);
      edge_scatter_kernel<<<E * 4, 256, 0, stream>>>(sp, dp, eedge, mb, sbuf, feat, acc, E);
    }
    {
      size_t n = (size_t)Nv * 4096;
      elu_div4_kernel<<<(unsigned)((n + 255) / 256), 256, 0, stream>>>(acc, hbuf, n);
      // ----- layer 2 init: acc = 4*h + sum_r b2 -----
      init_acc2_kernel<<<(unsigned)((n + 255) / 256), 256, 0, stream>>>(hbuf, bsum2, acc, n);
    }
    for (int r = 0; r < 4; ++r) {
      const int* sp = srcg[g] + (size_t)r * E;
      const int* dp = dstg[g] + (size_t)r * E;
      gemm_bf16_wmma<0><<<ggrid, 256, 0, stream>>>(hbuf, W2 + (size_t)r * 4096 * 4096,
                                                   feat, nullptr, Nv, 4096, 4096);
      el_er_kernel<<<Nv, 256, 0, stream>>>(feat, al2 + (size_t)r * 4096, ar2 + (size_t)r * 4096, elb, erb);
      init_ms_kernel<<<(Nv * 4 + 255) / 256, 256, 0, stream>>>(mb, sbuf, Nv * 4);
      edge_logits_kernel<<<(E * 4 + 255) / 256, 256, 0, stream>>>(sp, dp, elb, erb, eedge, mb, E);
      fix_m_kernel<<<(Nv * 4 + 255) / 256, 256, 0, stream>>>(mb, Nv * 4);
      edge_exp_kernel<<<(E * 4 + 255) / 256, 256, 0, stream>>>(dp, eedge, mb, sbuf, E);
      edge_scatter_kernel<<<E * 4, 256, 0, stream>>>(sp, dp, eedge, mb, sbuf, feat, acc, E);
    }
    col_mean_kernel<<<16, 256, 0, stream>>>(acc, repr + (size_t)g * 4096, Nv);
  }

  // classifier head
  gemv_kernel<<<(3072 + 255) / 256, 256, 0, stream>>>(repr, Wp, bp, zb, 12288, 3072, 0);
  gemv_kernel<<<4, 256, 0, stream>>>(zb, Wc1, bc1, hcb, 3072, 1024, 1);
  gemv_kernel<<<1, 256, 0, stream>>>(hcb, Wc2, bc2, (float*)d_out, 1024, 3, 0);
}
